// DecoderRNN_44933947851352
// MI455X (gfx1250) — matrix operand order repaired and checked by reading, not compiled, not run on hardware
//
#include <hip/hip_runtime.h>
#include <hip/hip_bf16.h>
#include <math.h>

#define EMBED   1024
#define HIDDEN  1024
#define VOCAB   32000
#define TSTEPS  64
#define KDIM    1024
#define GATES   4096

typedef __attribute__((ext_vector_type(16))) __bf16   v16bf;
typedef __attribute__((ext_vector_type(8)))  float    v8f;
typedef __attribute__((ext_vector_type(8)))  unsigned v8u;

// ---- pack hi16(x), hi16(y) -> one dword: exactly one V_PERM_B32 ------------
// bf16 truncation (round-toward-zero) of two f32 values per instruction.
static __device__ __forceinline__ unsigned pk_bf16_trunc(float x, float y) {
  // result byte0,1 = x bytes 2,3 (sel 2,3 from S1);  byte2,3 = y bytes 2,3 (sel 6,7 from S0)
  return __builtin_amdgcn_perm(__float_as_uint(y), __float_as_uint(x), 0x07060302u);
}

// scalar f32 -> bf16 RNE (cold kernels only)
static __device__ __forceinline__ __bf16 f2bf(float f) {
  unsigned u = __float_as_uint(f);
  u += 0x7FFFu + ((u >> 16) & 1u);
  unsigned short hh = (unsigned short)(u >> 16);
  return __builtin_bit_cast(__bf16, hh);
}

// B fragment: 16 contiguous f32 of one weight row -> 16 packed bf16 (8 perms)
static __device__ __forceinline__ v16bf load_b_frag(const float* __restrict__ p) {
  const float4* q = (const float4*)p;
  v8u w;
#pragma unroll
  for (int i = 0; i < 4; ++i) {
    float4 f = q[i];
    w[2 * i + 0] = pk_bf16_trunc(f.x, f.y);
    w[2 * i + 1] = pk_bf16_trunc(f.z, f.w);
  }
  return __builtin_bit_cast(v16bf, w);
}

// A fragment: two 16B chunks of pre-converted bf16, dword-level combine
static __device__ __forceinline__ v16bf load_a_frag(const __bf16* __restrict__ p) {
  uint4 lo = *(const uint4*)(p);
  uint4 hi = *(const uint4*)(p + 8);
  v8u w;
  w[0] = lo.x; w[1] = lo.y; w[2] = lo.z; w[3] = lo.w;
  w[4] = hi.x; w[5] = hi.y; w[6] = hi.z; w[7] = hi.w;
  return __builtin_bit_cast(v16bf, w);
}

// ---------------------------------------------------------------------------
// Prep: xs (features at t=0, embed[captions[t-1]] after) as bf16,
// fused bias, zeroed h/c state.
// ---------------------------------------------------------------------------
__global__ void prep_kernel(const int* __restrict__ captions,
                            const float* __restrict__ features,
                            const float* __restrict__ embed_table,
                            const float* __restrict__ b_ih,
                            const float* __restrict__ b_hh,
                            __bf16* __restrict__ Xbf,
                            float* __restrict__ bias,
                            float* __restrict__ h,
                            float* __restrict__ c) {
  int tid = blockIdx.x * blockDim.x + threadIdx.x;   // 65536 threads
  if (tid < TSTEPS * EMBED) {
    int t = tid >> 10, e = tid & (EMBED - 1);
    float v = (t == 0) ? features[e]
                       : embed_table[(size_t)captions[t - 1] * EMBED + e];
    Xbf[tid] = f2bf(v);
  }
  if (tid < GATES)  bias[tid] = b_ih[tid] + b_hh[tid];
  if (tid < HIDDEN) { h[tid] = 0.0f; c[tid] = 0.0f; }
}

// ---------------------------------------------------------------------------
// C[64,N] = A_bf16[64,K] @ B_f32[N,K]^T + bias[N]
// 8 waves/block; wave = (mt, nsub). Each wave: one 16-row tile x FOUR 16-col
// tiles sharing a single A fragment per K-step (4 WMMAs per A load; reuse_a
// hint chained). Block covers 64 rows x 128 cols; grid = N/128.
// ---------------------------------------------------------------------------
__global__ void __launch_bounds__(256)
gemm_bf16_wmma(const __bf16* __restrict__ A,
               const float*  __restrict__ B,
               const float*  __restrict__ bias,
               float* __restrict__ C,
               int N) {
  const int lane = threadIdx.x & 31;
  const int wave = threadIdx.x >> 5;                 // 0..7
  const int mt   = wave >> 1;                        // 0..3 (16-row tile)
  const int half = lane >> 4;                        // 0 or 1
  const int l15  = lane & 15;

  const int arow = mt * 16 + l15;
  const int ncol = blockIdx.x * 128 + (wave & 1) * 64 + l15; // first of 4 tiles

  const __bf16* Ap = A + (size_t)arow * KDIM + half * 16;
  const float*  Bp0 = B + (size_t)(ncol +  0) * KDIM + half * 16;
  const float*  Bp1 = B + (size_t)(ncol + 16) * KDIM + half * 16;
  const float*  Bp2 = B + (size_t)(ncol + 32) * KDIM + half * 16;
  const float*  Bp3 = B + (size_t)(ncol + 48) * KDIM + half * 16;

  v8f acc0, acc1, acc2, acc3;
#pragma unroll
  for (int i = 0; i < 8; ++i) {
    acc0[i] = 0.0f; acc1[i] = 0.0f; acc2[i] = 0.0f; acc3[i] = 0.0f;
  }

  for (int kk = 0; kk < KDIM; kk += 32) {
    v16bf a  = load_a_frag(Ap + kk);
    __builtin_prefetch(Bp0 + kk + 32, 0, 1);
    __builtin_prefetch(Bp1 + kk + 32, 0, 1);
    __builtin_prefetch(Bp2 + kk + 32, 0, 1);
    __builtin_prefetch(Bp3 + kk + 32, 0, 1);
    v16bf b0 = load_b_frag(Bp0 + kk);
    v16bf b1 = load_b_frag(Bp1 + kk);
    v16bf b2 = load_b_frag(Bp2 + kk);
    v16bf b3 = load_b_frag(Bp3 + kk);
    // back-to-back identical WMMAs sharing A: reuse_a chained on first three
    acc0 = __builtin_amdgcn_wmma_f32_16x16x32_bf16(
        false, a, false, b0, (short)0, acc0, /*reuse_a=*/true,  false);
    acc1 = __builtin_amdgcn_wmma_f32_16x16x32_bf16(
        false, a, false, b1, (short)0, acc1, /*reuse_a=*/true,  false);
    acc2 = __builtin_amdgcn_wmma_f32_16x16x32_bf16(
        false, a, false, b2, (short)0, acc2, /*reuse_a=*/true,  false);
    acc3 = __builtin_amdgcn_wmma_f32_16x16x32_bf16(
        false, a, false, b3, (short)0, acc3, /*reuse_a=*/false, false);
  }

  const float bb0 = bias[ncol +  0];
  const float bb1 = bias[ncol + 16];
  const float bb2 = bias[ncol + 32];
  const float bb3 = bias[ncol + 48];
#pragma unroll
  for (int v = 0; v < 8; ++v) {
    int row = mt * 16 + v + half * 8;   // D layout: lanes16-31 hold M=+8
    float* Crow = C + (size_t)row * N + ncol;
    Crow[ 0] = acc0[v] + bb0;
    Crow[16] = acc1[v] + bb1;
    Crow[32] = acc2[v] + bb2;
    Crow[48] = acc3[v] + bb3;
  }
}

// ---------------------------------------------------------------------------
// Per-step GEMV: gates[j] = Gin_t[j] + dot(W_hh[j,:], h)   (one wave per row)
// ---------------------------------------------------------------------------
__global__ void __launch_bounds__(256)
lstm_gates_kernel(const float* __restrict__ Whh,
                  const float* __restrict__ h,
                  const float* __restrict__ Gin_t,
                  float* __restrict__ gates) {
  const int lane = threadIdx.x & 31;
  const int j    = blockIdx.x * 8 + (threadIdx.x >> 5);
  const float4* wr = (const float4*)(Whh + (size_t)j * HIDDEN);
  const float4* hr = (const float4*)h;
  float acc = 0.0f;
#pragma unroll
  for (int it = 0; it < 8; ++it) {
    float4 w = wr[it * 32 + lane];
    float4 x = hr[it * 32 + lane];
    acc += w.x * x.x + w.y * x.y + w.z * x.z + w.w * x.w;
  }
#pragma unroll
  for (int off = 16; off > 0; off >>= 1) acc += __shfl_xor(acc, off, 32);
  if (lane == 0) gates[j] = Gin_t[j] + acc;
}

// ---------------------------------------------------------------------------
// Per-step elementwise LSTM cell update; emits h in bf16 for the FC GEMM.
// ---------------------------------------------------------------------------
__global__ void lstm_update_kernel(const float* __restrict__ gates,
                                   float* __restrict__ h,
                                   float* __restrict__ c,
                                   __bf16* __restrict__ hs_bf,
                                   int t) {
  int e = blockIdx.x * blockDim.x + threadIdx.x;
  if (e >= HIDDEN) return;
  float ig = 1.0f / (1.0f + __expf(-gates[e]));
  float fg = 1.0f / (1.0f + __expf(-gates[HIDDEN + e]));
  float gg = tanhf(gates[2 * HIDDEN + e]);
  float og = 1.0f / (1.0f + __expf(-gates[3 * HIDDEN + e]));
  float cn = fg * c[e] + ig * gg;
  float hn = og * tanhf(cn);
  c[e] = cn;
  h[e] = hn;
  hs_bf[(size_t)t * HIDDEN + e] = f2bf(hn);
}

// ---------------------------------------------------------------------------
extern "C" void kernel_launch(void* const* d_in, const int* in_sizes, int n_in,
                              void* d_out, int out_size, void* d_ws, size_t ws_size,
                              hipStream_t stream) {
  (void)in_sizes; (void)n_in; (void)out_size; (void)ws_size;

  const int*   captions    = (const int*)  d_in[0];
  const float* features    = (const float*)d_in[1];
  const float* embed_table = (const float*)d_in[2];
  const float* W_ih        = (const float*)d_in[3];
  const float* W_hh        = (const float*)d_in[4];
  const float* b_ih        = (const float*)d_in[5];
  const float* b_hh        = (const float*)d_in[6];
  const float* fc_W        = (const float*)d_in[7];
  const float* fc_b        = (const float*)d_in[8];
  float* logits = (float*)d_out;

  // workspace layout (256B aligned slabs)
  char* ws = (char*)d_ws;
  size_t off = 0;
  auto alloc = [&](size_t bytes) {
    char* p = ws + off;
    off += (bytes + 255) & ~(size_t)255;
    return p;
  };
  __bf16* Xbf   = (__bf16*)alloc((size_t)TSTEPS * EMBED * sizeof(__bf16));   // 128 KB
  float*  Gin   = (float*) alloc((size_t)TSTEPS * GATES * sizeof(float));    // 1 MB
  float*  bias  = (float*) alloc(GATES * sizeof(float));
  float*  hvec  = (float*) alloc(HIDDEN * sizeof(float));
  float*  cvec  = (float*) alloc(HIDDEN * sizeof(float));
  float*  gates = (float*) alloc(GATES * sizeof(float));
  __bf16* hsbf  = (__bf16*)alloc((size_t)TSTEPS * HIDDEN * sizeof(__bf16));  // 128 KB

  // 1) embedding gather + input staging + bias fuse + state zeroing
  prep_kernel<<<(TSTEPS * EMBED + 255) / 256, 256, 0, stream>>>(
      captions, features, embed_table, b_ih, b_hh, Xbf, bias, hvec, cvec);

  // 2) hoisted input-side GEMM: Gin[64,4096] = X @ W_ih^T + (b_ih + b_hh)
  gemm_bf16_wmma<<<GATES / 128, 256, 0, stream>>>(Xbf, W_ih, bias, Gin, GATES);

  // 3) recurrence: stream ordering provides the t -> t+1 dependency
  for (int t = 0; t < TSTEPS; ++t) {
    lstm_gates_kernel<<<GATES / 8, 256, 0, stream>>>(
        W_hh, hvec, Gin + (size_t)t * GATES, gates);
    lstm_update_kernel<<<HIDDEN / 256, 256, 0, stream>>>(
        gates, hvec, cvec, hsbf, t);
  }

  // 4) FC GEMM: logits[64,32000] = hs @ fc_W^T + fc_b (fc_W read once, ~131 MB)
  gemm_bf16_wmma<<<VOCAB / 128, 256, 0, stream>>>(hsbf, fc_W, fc_b, logits, VOCAB);
}
